// RetentiveNetwork_60258391162951
// MI455X (gfx1250) — matrix operand order repaired and checked by reading, not compiled, not run on hardware
//
#include <hip/hip_runtime.h>

// ---------------------------------------------------------------------------
// RetNet forward, CDNA5 (gfx1250), wave32, WMMA bf16 path.
// Round 4: f32->bf16 via native fptrunc casts (backend may select
// v_cvt_pk_bf16_f32); guarded builtin first; else identical to round 3.
// ---------------------------------------------------------------------------

#define HID   2048
#define NHEAD 8
#define DHEAD 256
#define FFD   8192
#define BATCH 4
#define SEQ   1024
#define NTOK  (BATCH * SEQ)          // 4096

typedef __bf16  bf16_t;
typedef bf16_t  v16bf __attribute__((ext_vector_type(16)));
typedef bf16_t  v2bf  __attribute__((ext_vector_type(2)));
typedef float   v8f   __attribute__((ext_vector_type(8)));
typedef float   v4f   __attribute__((ext_vector_type(4)));
typedef unsigned int v4u __attribute__((ext_vector_type(4)));
typedef unsigned int v2u __attribute__((ext_vector_type(2)));

union FragBF {
    v16bf v;
    v4u   q[2];
    unsigned int u[8];
};

#if __has_builtin(__builtin_amdgcn_cvt_pk_bf16_f32)
#define HW_PK_BF16 1
#else
#define HW_PK_BF16 0
#endif

__device__ __forceinline__ unsigned int pack2bf(float a, float b) {
#if HW_PK_BF16
    auto r = __builtin_amdgcn_cvt_pk_bf16_f32(a, b);  // v_cvt_pk_bf16_f32 (RNE)
    unsigned int u;
    __builtin_memcpy(&u, &r, 4);
    return u;
#else
    // plain fptrunc: backend selects native bf16 convert if the target has it
    v2bf t;
    t.x = (bf16_t)a;
    t.y = (bf16_t)b;
    unsigned int u;
    __builtin_memcpy(&u, &t, 4);
    return u;
#endif
}
__device__ __forceinline__ unsigned short f2bf(float f) {
    bf16_t h = (bf16_t)f;
    unsigned short u;
    __builtin_memcpy(&u, &h, 2);
    return u;
}

__device__ __forceinline__ v8f wmma_bf16(v16bf a, v16bf b, v8f c) {
    // v_wmma_f32_16x16x32_bf16  (neg_a, A, neg_b, B, c_mod, C, reuse_a, reuse_b)
    return __builtin_amdgcn_wmma_f32_16x16x32_bf16(false, a, false, b, (short)0, c,
                                                   false, false);
}

// A-fragment (16x32 bf16) from LDS, row-major tile, stride in halves.
// Contiguous-k layout used here: lanes 0-15 row M=lane, K = kh*16..kh*16+15, kh=lane>>4.
__device__ __forceinline__ v16bf load_a_lds(const unsigned short* s, int row0,
                                            int k0, int stride) {
    int lane = threadIdx.x & 31;
    int m  = lane & 15;
    int kh = lane >> 4;
    const unsigned short* p = s + (row0 + m) * stride + k0;
    FragBF f;
    f.q[0] = *(const v4u*)(p + kh * 16);
    f.q[1] = *(const v4u*)(p + kh * 16 + 8);
    return f.v;
}

// B-fragment (32x16 bf16) from LDS holding B transposed as [n][k].
// Contiguous-k layout used here: lanes 0-15 col N=lane hold K=0..15; lanes 16-31 hold K=16..31.
__device__ __forceinline__ v16bf load_b_lds(const unsigned short* s, int n0,
                                            int k0, int stride) {
    int lane = threadIdx.x & 31;
    int n  = lane & 15;
    int kh = lane >> 4;
    const unsigned short* p = s + (n0 + n) * stride + k0 + kh * 16;
    FragBF f;
    f.q[0] = *(const v4u*)(p);
    f.q[1] = *(const v4u*)(p + 8);
    return f.v;
}

// A-fragment straight from global f32 (row-major, stride in floats), f32->bf16.
__device__ __forceinline__ v16bf load_a_gf32(const float* g, int row0, int k0,
                                             long stride) {
    int lane = threadIdx.x & 31;
    int m  = lane & 15;
    int kh = lane >> 4;
    const float* p = g + (long)(row0 + m) * stride + k0;
    v4f a0 = *(const v4f*)(p + kh * 16);
    v4f a1 = *(const v4f*)(p + kh * 16 + 4);
    v4f a2 = *(const v4f*)(p + kh * 16 + 8);
    v4f a3 = *(const v4f*)(p + kh * 16 + 12);
    FragBF f;
    f.u[0] = pack2bf(a0.x, a0.y);  f.u[1] = pack2bf(a0.z, a0.w);
    f.u[2] = pack2bf(a1.x, a1.y);  f.u[3] = pack2bf(a1.z, a1.w);
    f.u[4] = pack2bf(a2.x, a2.y);  f.u[5] = pack2bf(a2.z, a2.w);
    f.u[6] = pack2bf(a3.x, a3.y);  f.u[7] = pack2bf(a3.z, a3.w);
    return f.v;
}

// B-fragment from global where B[k][n] = g[n][k] (i.e. we multiply by g^T):
// per lane fixed n, contiguous 16 k values -> contiguous in g. f32->bf16.
__device__ __forceinline__ v16bf load_bT_gf32(const float* g, int n0, int k0,
                                              long stride) {
    int lane = threadIdx.x & 31;
    int n  = lane & 15;
    int kh = lane >> 4;
    const float* p = g + (long)(n0 + n) * stride + k0 + kh * 16;
    v4f a0 = *(const v4f*)(p);
    v4f a1 = *(const v4f*)(p + 4);
    v4f a2 = *(const v4f*)(p + 8);
    v4f a3 = *(const v4f*)(p + 12);
    FragBF f;
    f.u[0] = pack2bf(a0.x, a0.y);  f.u[1] = pack2bf(a0.z, a0.w);
    f.u[2] = pack2bf(a1.x, a1.y);  f.u[3] = pack2bf(a1.z, a1.w);
    f.u[4] = pack2bf(a2.x, a2.y);  f.u[5] = pack2bf(a2.z, a2.w);
    f.u[6] = pack2bf(a3.x, a3.y);  f.u[7] = pack2bf(a3.z, a3.w);
    return f.v;
}

// branch-free tanh via exp: tanh(y) = 1 - 2/(exp(2y)+1); saturates correctly.
__device__ __forceinline__ float fast_tanh(float y) {
    float e = __expf(2.f * y);
    return 1.f - 2.f / (e + 1.f);
}
__device__ __forceinline__ float gelu_tanh(float x) {
    float u = 0.79788456080286535588f * (x + 0.044715f * x * x * x);
    return 0.5f * x * (1.f + fast_tanh(u));
}

// ---------------------------------------------------------------------------
// Generic GEMM: C[M,N] = epi(A[M,K] @ B[K,N]); fp32 in/out, bf16 WMMA inside.
// Double-buffered LDS pipeline: stage tile t+1 from global into registers
// while WMMAs consume tile t from LDS; one barrier per K-step.
// ---------------------------------------------------------------------------
#define BM 128
#define BN 128
#define BK 32
#define LDT 40   // LDS row stride in halves (+8 pad, keeps 16B alignment)

#define EPI_NONE      0
#define EPI_ADD       1
#define EPI_BIAS      2
#define EPI_BIAS_GELU 3
#define EPI_BIAS_ADD  4

__global__ __launch_bounds__(256) void gemm_kernel(
    const float* __restrict__ A, const float* __restrict__ B,
    float* __restrict__ C, int M, int N, int K,
    const float* __restrict__ bias, const float* __restrict__ res, int epi) {
    __shared__ unsigned short As[2][BM * LDT];
    __shared__ unsigned short Bs[2][BN * LDT];   // B transposed: [n][k]

    int tid  = threadIdx.x;
    int bm   = blockIdx.y * BM;
    int bn   = blockIdx.x * BN;
    int wave = tid >> 5;
    int lane = tid & 31;
    int wm   = (wave >> 1) * 32;   // wave row offset inside block tile
    int wn   = (wave & 1) * 64;    // wave col offset inside block tile

    // staging thread mapping
    int ar = tid >> 3;             // A row (0..31), +32 per pass
    int ac = (tid & 7) * 4;        // A col (0..28)
    int bk = tid >> 5;             // B k-row (0..7), +8 per pass
    int bc = (tid & 31) * 4;       // B col (0..124)

    v8f acc[2][4] = {};
    v4f avreg[4], bvreg[4];

    // ---- preload tile 0 into registers, then LDS buffer 0 ----
    #pragma unroll
    for (int p = 0; p < 4; ++p)
        avreg[p] = *(const v4f*)(A + (long)(bm + ar + p * 32) * K + ac);
    #pragma unroll
    for (int p = 0; p < 4; ++p)
        bvreg[p] = *(const v4f*)(B + (long)(bk + p * 8) * N + bn + bc);
    #pragma unroll
    for (int p = 0; p < 4; ++p) {
        v2u t; t.x = pack2bf(avreg[p].x, avreg[p].y);
        t.y = pack2bf(avreg[p].z, avreg[p].w);
        *(v2u*)&As[0][(ar + p * 32) * LDT + ac] = t;
    }
    #pragma unroll
    for (int p = 0; p < 4; ++p) {
        int krow = bk + p * 8;
        Bs[0][(bc + 0) * LDT + krow] = f2bf(bvreg[p].x);
        Bs[0][(bc + 1) * LDT + krow] = f2bf(bvreg[p].y);
        Bs[0][(bc + 2) * LDT + krow] = f2bf(bvreg[p].z);
        Bs[0][(bc + 3) * LDT + krow] = f2bf(bvreg[p].w);
    }
    __syncthreads();

    int nk = K / BK;
    for (int t = 0; t < nk; ++t) {
        int cur = t & 1;
        bool more = (t + 1) < nk;

        // ---- issue global loads for tile t+1 (overlaps with WMMAs below) ----
        if (more) {
            int k0 = (t + 1) * BK;
            if (t + 2 < nk) {   // deep prefetch into GL2 (global_prefetch_b8)
                __builtin_prefetch(A + (long)(bm + ar) * K + (t + 2) * BK + ac, 0, 1);
                __builtin_prefetch(B + (long)((t + 2) * BK + bk) * N + bn + bc, 0, 1);
            }
            #pragma unroll
            for (int p = 0; p < 4; ++p)
                avreg[p] = *(const v4f*)(A + (long)(bm + ar + p * 32) * K + k0 + ac);
            #pragma unroll
            for (int p = 0; p < 4; ++p)
                bvreg[p] = *(const v4f*)(B + (long)(k0 + bk + p * 8) * N + bn + bc);
        }

        // ---- compute on tile t from LDS ----
        v16bf af[2], bf[4];
        af[0] = load_a_lds(As[cur], wm, 0, LDT);
        af[1] = load_a_lds(As[cur], wm + 16, 0, LDT);
        #pragma unroll
        for (int j = 0; j < 4; ++j) bf[j] = load_b_lds(Bs[cur], wn + j * 16, 0, LDT);
        #pragma unroll
        for (int i = 0; i < 2; ++i)
            #pragma unroll
            for (int j = 0; j < 4; ++j)
                acc[i][j] = wmma_bf16(af[i], bf[j], acc[i][j]);

        // ---- commit tile t+1 to the other LDS buffer ----
        if (more) {
            int nxt = cur ^ 1;
            #pragma unroll
            for (int p = 0; p < 4; ++p) {
                v2u tt; tt.x = pack2bf(avreg[p].x, avreg[p].y);
                tt.y = pack2bf(avreg[p].z, avreg[p].w);
                *(v2u*)&As[nxt][(ar + p * 32) * LDT + ac] = tt;
            }
            #pragma unroll
            for (int p = 0; p < 4; ++p) {
                int krow = bk + p * 8;
                Bs[nxt][(bc + 0) * LDT + krow] = f2bf(bvreg[p].x);
                Bs[nxt][(bc + 1) * LDT + krow] = f2bf(bvreg[p].y);
                Bs[nxt][(bc + 2) * LDT + krow] = f2bf(bvreg[p].z);
                Bs[nxt][(bc + 3) * LDT + krow] = f2bf(bvreg[p].w);
            }
        }
        __syncthreads();
    }

    // ---- epilogue ----
    bool has_bias = (epi == EPI_BIAS) || (epi == EPI_BIAS_GELU) || (epi == EPI_BIAS_ADD);
    bool has_res  = (epi == EPI_ADD) || (epi == EPI_BIAS_ADD);
    bool do_gelu  = (epi == EPI_BIAS_GELU);
    int lr = (lane >> 4) * 8;   // C layout: VGPR r -> M = lr + r
    int lc = lane & 15;         //           N = lane & 15
    #pragma unroll
    for (int i = 0; i < 2; ++i) {
        #pragma unroll
        for (int j = 0; j < 4; ++j) {
            int col = bn + wn + j * 16 + lc;
            float bv = has_bias ? bias[col] : 0.f;
            #pragma unroll
            for (int r = 0; r < 8; ++r) {
                int row = bm + wm + i * 16 + lr + r;
                float v = acc[i][j][r] + bv;
                if (do_gelu) v = gelu_tanh(v);
                if (has_res) v += res[(long)row * N + col];
                C[(long)row * N + col] = v;
            }
        }
    }
}

// ---------------------------------------------------------------------------
// Fused retention: per (b,h,n-tile of 64) compute Y = (QK^T * D) @ V without
// materializing the S x S score matrix. Q frags resident in VGPRs.
// ---------------------------------------------------------------------------
#define RLDP 72   // LDS stride in halves for 64-wide / transposed-V tiles

__global__ __launch_bounds__(256) void retention_kernel(
    const float* __restrict__ Q, const float* __restrict__ Kmat,
    const float* __restrict__ V, float* __restrict__ Y) {
    __shared__ unsigned short Ps[64 * RLDP];    // decayed scores tile, bf16
    __shared__ unsigned short Vt[DHEAD * RLDP]; // V tile transposed [d][m], bf16

    int bh = blockIdx.y;                 // b * NHEAD + h
    int h  = bh & (NHEAD - 1);
    long headoff = (long)(bh >> 3) * ((long)SEQ * HID) + (long)h * DHEAD;
    int n0   = blockIdx.x * 64;
    int tid  = threadIdx.x;
    int wave = tid >> 5;
    int lane = tid & 31;
    int wi = wave >> 1;                  // 0..3: 16-row group
    int wj = wave & 1;                   // 0..1: col half (128 d cols / 32 m cols)

    // per-head decay: gamma_h = 1 - exp(linspace(ln(1/32), ln(1/512), 8)[h])
    const float l0 = -3.46573590279972654709f;   // ln(1/32)
    const float l1 = -6.23832462503951057771f;   // ln(1/512)
    float lg = logf(1.f - __expf(l0 + (l1 - l0) * (float)h * (1.f / 7.f)));

    // resident Q fragments: rows n0 + wi*16 .. +15, all 256 d (8 frags of K=32)
    const float* Qb = Q + headoff;
    v16bf qf[8];
    #pragma unroll
    for (int dk = 0; dk < 8; ++dk)
        qf[dk] = load_a_gf32(Qb, n0 + wi * 16, dk * 32, HID);

    v8f yacc[8] = {};
    const float* Kb = Kmat + headoff;
    const float* Vb = V + headoff;
    int lr = (lane >> 4) * 8;
    int lc = lane & 15;

    for (int mt = 0; mt <= blockIdx.x; ++mt) {
        int m0 = mt * 64;

        // prefetch next m-tile's K and V rows into GL2
        if (mt < blockIdx.x) {
            int pm = (tid & 63);
            int pd = (tid >> 6) * 64;
            __builtin_prefetch(Kb + (long)(m0 + 64 + pm) * HID + pd, 0, 1);
            __builtin_prefetch(Vb + (long)(m0 + 64 + pm) * HID + pd, 0, 1);
        }

        // ---- stage V tile [64m x 256d] transposed into Vt[d][m] ----
        {
            int m  = tid & 63;
            int d0 = (tid >> 6) * 64;
            const float* vp = Vb + (long)(m0 + m) * HID + d0;
            #pragma unroll
            for (int q = 0; q < 16; ++q) {
                v4f vv = *(const v4f*)(vp + q * 4);
                int d = d0 + q * 4;
                Vt[(d + 0) * RLDP + m] = f2bf(vv.x);
                Vt[(d + 1) * RLDP + m] = f2bf(vv.y);
                Vt[(d + 2) * RLDP + m] = f2bf(vv.z);
                Vt[(d + 3) * RLDP + m] = f2bf(vv.w);
            }
        }

        // ---- stage 1: P[64x64] = Q @ K^T, K frags straight from global ----
        v8f pacc[2] = {};
        #pragma unroll
        for (int dk = 0; dk < 8; ++dk) {
            v16bf b0 = load_bT_gf32(Kb, m0 + wj * 32, dk * 32, HID);
            v16bf b1 = load_bT_gf32(Kb, m0 + wj * 32 + 16, dk * 32, HID);
            pacc[0] = wmma_bf16(qf[dk], b0, pacc[0]);
            pacc[1] = wmma_bf16(qf[dk], b1, pacc[1]);
        }
        // decay mask gamma^(n-m), causal; write bf16 to LDS
        #pragma unroll
        for (int t = 0; t < 2; ++t) {
            #pragma unroll
            for (int r = 0; r < 8; ++r) {
                int n = n0 + wi * 16 + lr + r;
                int m = m0 + wj * 32 + t * 16 + lc;
                int p = n - m;
                float v = (p >= 0) ? pacc[t][r] * __expf((float)p * lg) : 0.f;
                Ps[(wi * 16 + lr + r) * RLDP + wj * 32 + t * 16 + lc] = f2bf(v);
            }
        }
        __syncthreads();

        // ---- stage 2: Y[64x256] += P @ V ----
        #pragma unroll
        for (int kc = 0; kc < 2; ++kc) {
            v16bf af = load_a_lds(Ps, wi * 16, kc * 32, RLDP);
            #pragma unroll
            for (int j = 0; j < 8; ++j) {
                v16bf bfr = load_b_lds(Vt, wj * 128 + j * 16, kc * 32, RLDP);
                yacc[j] = wmma_bf16(af, bfr, yacc[j]);
            }
        }
        __syncthreads();
    }

    // ---- store Y ----
    float* Yb = Y + headoff;
    #pragma unroll
    for (int j = 0; j < 8; ++j) {
        int col = wj * 128 + j * 16 + lc;
        #pragma unroll
        for (int r = 0; r < 8; ++r) {
            int row = n0 + wi * 16 + lr + r;
            Yb[(long)row * HID + col] = yacc[j][r];
        }
    }
}

// ---------------------------------------------------------------------------
// LayerNorm over H=2048 per token. 256 threads, 8 channels each.
// ---------------------------------------------------------------------------
__device__ __forceinline__ float block_reduce_sum(float v, float* red) {
    int tid = threadIdx.x;
    red[tid] = v;
    __syncthreads();
    for (int s = 128; s > 0; s >>= 1) {
        if (tid < s) red[tid] += red[tid + s];
        __syncthreads();
    }
    float r = red[0];
    __syncthreads();
    return r;
}

__global__ __launch_bounds__(256) void layernorm_kernel(
    const float* __restrict__ X, const float* __restrict__ g,
    const float* __restrict__ b, float* __restrict__ O) {
    __shared__ float red[256];
    long row = blockIdx.x;
    const float* x = X + row * HID;
    float* o = O + row * HID;
    int tid = threadIdx.x;
    v4f a = *(const v4f*)(x + tid * 8);
    v4f c = *(const v4f*)(x + tid * 8 + 4);
    float s  = a.x + a.y + a.z + a.w + c.x + c.y + c.z + c.w;
    float sq = a.x * a.x + a.y * a.y + a.z * a.z + a.w * a.w +
               c.x * c.x + c.y * c.y + c.z * c.z + c.w * c.w;
    float sum  = block_reduce_sum(s, red);
    float sums = block_reduce_sum(sq, red);
    float mu  = sum * (1.f / HID);
    float var = sums * (1.f / HID) - mu * mu;
    float rs  = rsqrtf(var + 1e-5f);
    #pragma unroll
    for (int k = 0; k < 8; ++k) {
        int ch = tid * 8 + k;
        float v = (k < 4) ? ((const float*)&a)[k] : ((const float*)&c)[k - 4];
        o[ch] = (v - mu) * rs * g[ch] + b[ch];
    }
}

// ---------------------------------------------------------------------------
// GroupNorm(HEADS groups) + SiLU gate: Z = silu(G) * (GN(Y)*gamma + beta).
// One block per (token, head); 256 threads = HEAD channels.
// ---------------------------------------------------------------------------
__global__ __launch_bounds__(256) void gngate_kernel(
    const float* __restrict__ Y, const float* __restrict__ G,
    const float* __restrict__ gg, const float* __restrict__ gb,
    float* __restrict__ Z) {
    __shared__ float red[256];
    int t = blockIdx.x;                 // token * NHEAD + head
    int head = t & (NHEAD - 1);
    long tok = (long)(t >> 3);
    long base = tok * HID + (long)head * DHEAD;
    int d = threadIdx.x;
    float y = Y[base + d];
    float mu = block_reduce_sum(y, red) * (1.f / DHEAD);
    float dy = y - mu;
    float var = block_reduce_sum(dy * dy, red) * (1.f / DHEAD);
    float yn = dy * rsqrtf(var + 1e-5f);
    int ch = head * DHEAD + d;
    yn = yn * gg[ch] + gb[ch];
    float gv = G[base + d];
    float sig = 1.f / (1.f + __expf(-gv));
    Z[base + d] = gv * sig * yn;
}

// ---------------------------------------------------------------------------
// Launcher
// ---------------------------------------------------------------------------
extern "C" void kernel_launch(void* const* d_in, const int* in_sizes, int n_in,
                              void* d_out, int out_size, void* d_ws, size_t ws_size,
                              hipStream_t stream) {
    const float* x    = (const float*)d_in[0];
    const float* Wq   = (const float*)d_in[1];
    const float* Wk   = (const float*)d_in[2];
    const float* Wv   = (const float*)d_in[3];
    const float* Wg   = (const float*)d_in[4];
    const float* Wo   = (const float*)d_in[5];
    const float* gng  = (const float*)d_in[6];
    const float* gnb  = (const float*)d_in[7];
    const float* W1   = (const float*)d_in[8];
    const float* b1   = (const float*)d_in[9];
    const float* W2   = (const float*)d_in[10];
    const float* b2   = (const float*)d_in[11];
    const float* lng  = (const float*)d_in[12];
    const float* lnb  = (const float*)d_in[13];

    float* ws = (float*)d_ws;
    const long T = (long)NTOK * HID;    // 8,388,608 floats
    float* XN  = ws;                    // also reused for RN
    float* Qb  = ws + 1 * T;            // Q/K/V/G region, reused as H1 (4T)
    float* Kb  = ws + 2 * T;
    float* Vb  = ws + 3 * T;
    float* Gb  = ws + 4 * T;
    float* Yb  = ws + 5 * T;
    float* Zb  = ws + 6 * T;
    float* RET = ws + 7 * T;
    float* XB  = ws + 8 * T;
    float* H1  = Qb;                    // [NTOK, FFD] overlays Q..G after retention

    dim3 blk(256);
    dim3 gH(HID / BN, NTOK / BM);       // (16, 32)
    dim3 gF(FFD / BN, NTOK / BM);       // (64, 32)
    dim3 gln(NTOK);
    dim3 ggn(NTOK * NHEAD);
    dim3 gret(SEQ / 64, BATCH * NHEAD); // (16, 32)

    const float* xin = x;
    for (int l = 0; l < 2; ++l) {
        const float* Wq_l = Wq + (long)l * HID * HID;
        const float* Wk_l = Wk + (long)l * HID * HID;
        const float* Wv_l = Wv + (long)l * HID * HID;
        const float* Wg_l = Wg + (long)l * HID * HID;
        const float* Wo_l = Wo + (long)l * HID * HID;
        const float* W1_l = W1 + (long)l * HID * FFD;
        const float* W2_l = W2 + (long)l * FFD * HID;
        const float* b1_l = b1 + (long)l * FFD;
        const float* b2_l = b2 + (long)l * HID;
        const float* gg_l = gng + (long)l * HID;
        const float* gb_l = gnb + (long)l * HID;
        float* xout = (l == 0) ? XB : (float*)d_out;

        layernorm_kernel<<<gln, blk, 0, stream>>>(xin, lng, lnb, XN);
        gemm_kernel<<<gH, blk, 0, stream>>>(XN, Wq_l, Qb, NTOK, HID, HID,
                                            nullptr, nullptr, EPI_NONE);
        gemm_kernel<<<gH, blk, 0, stream>>>(XN, Wk_l, Kb, NTOK, HID, HID,
                                            nullptr, nullptr, EPI_NONE);
        gemm_kernel<<<gH, blk, 0, stream>>>(XN, Wv_l, Vb, NTOK, HID, HID,
                                            nullptr, nullptr, EPI_NONE);
        gemm_kernel<<<gH, blk, 0, stream>>>(XN, Wg_l, Gb, NTOK, HID, HID,
                                            nullptr, nullptr, EPI_NONE);
        retention_kernel<<<gret, blk, 0, stream>>>(Qb, Kb, Vb, Yb);
        gngate_kernel<<<ggn, blk, 0, stream>>>(Yb, Gb, gg_l, gb_l, Zb);
        gemm_kernel<<<gH, blk, 0, stream>>>(Zb, Wo_l, RET, NTOK, HID, HID,
                                            nullptr, xin, EPI_ADD);
        layernorm_kernel<<<gln, blk, 0, stream>>>(RET, lng, lnb, XN);
        gemm_kernel<<<gF, blk, 0, stream>>>(XN, W1_l, H1, NTOK, FFD, HID,
                                            b1_l, nullptr, EPI_BIAS_GELU);
        gemm_kernel<<<gH, blk, 0, stream>>>(H1, W2_l, xout, NTOK, HID, FFD,
                                            b2_l, RET, EPI_BIAS_ADD);
        xin = xout;
    }
}
